// Model_76811195122407
// MI455X (gfx1250) — matrix operand order repaired and checked by reading, not compiled, not run on hardware
//
#include <hip/hip_runtime.h>

// ---------------------------------------------------------------------------
// 3-layer LIF SNN, T=50, B=16384, D_in=D_out=128, H=100 (padded to 112/128).
// bf16 WMMA (v_wmma_f32_16x16x32_bf16) with f32 accumulate.
//  - Each wave owns 16 batch rows for all 50 steps; membranes stay in VGPRs.
//  - Layer-1 A fragments built straight from global x (no LDS round trip).
//  - A fragments preloaded once per layer and reused across all N tiles.
//  - All 4 B fragments of a tile loaded before the 4 chained WMMAs.
// ---------------------------------------------------------------------------

typedef __attribute__((ext_vector_type(16))) __bf16 v16bf;
typedef __attribute__((ext_vector_type(8)))  __bf16 v8bf;
typedef __attribute__((ext_vector_type(8)))  float  v8f;

#define T_STEPS 50
#define BATCH   16384
#define DIM     128      // D_in = D_out
#define H_REAL  100
#define HP      112      // H padded to 7 WMMA column tiles
#define NT_H    7        // N tiles for layers 1,2
#define NT_O    8        // N tiles for layer 3
#define WAVES   4
#define ROWS_PER_WAVE 16

#define ROWB    256      // bf16 row stride in bytes (128 cols * 2B)
#define LDS_W1  0
#define LDS_W2  (HP * ROWB)              // 28672
#define LDS_W3  (LDS_W2 + HP * ROWB)     // 57344
#define LDS_WAVE0 (LDS_W3 + 128 * ROWB)  // 90112
#define WAVE_LDS 8192                    // spkA(4K) + spkB(4K)
#define LDS_TOTAL (LDS_WAVE0 + WAVES * WAVE_LDS)  // 122880

#define BETA 0.9f
#define THR  0.5f

// A fragment (16x32 bf16, ISA 7.12.2): lane L -> row L%16;
// lanes 0-15: K = k0+0..7 (VGPR0-3), k0+16..23 (VGPR4-7); lanes 16-31: +8.
__device__ __forceinline__ v16bf load_fragA_lds(const char* buf, int k0, int lane) {
  const char* p = buf + (lane & 15) * ROWB + k0 * 2 + (lane >> 4) * 16;
  v8bf c0 = *(const v8bf*)(p);
  v8bf c1 = *(const v8bf*)(p + 32);
  return __builtin_shufflevector(c0, c1, 0,1,2,3,4,5,6,7,8,9,10,11,12,13,14,15);
}

// Same A fragment built directly from global f32 x (row stride = 128 floats),
// converted to bf16 in flight. Each half is 8 contiguous floats (two b128s).
__device__ __forceinline__ v16bf load_fragA_global(const float* xbase, int k0, int lane) {
  const float* p = xbase + (size_t)(lane & 15) * DIM + k0 + (lane >> 4) * 8;
  float4 f0 = *(const float4*)(p);
  float4 f1 = *(const float4*)(p + 4);
  float4 f2 = *(const float4*)(p + 16);
  float4 f3 = *(const float4*)(p + 20);
  v16bf r;
  r[0]=(__bf16)f0.x;  r[1]=(__bf16)f0.y;  r[2]=(__bf16)f0.z;  r[3]=(__bf16)f0.w;
  r[4]=(__bf16)f1.x;  r[5]=(__bf16)f1.y;  r[6]=(__bf16)f1.z;  r[7]=(__bf16)f1.w;
  r[8]=(__bf16)f2.x;  r[9]=(__bf16)f2.y;  r[10]=(__bf16)f2.z; r[11]=(__bf16)f2.w;
  r[12]=(__bf16)f3.x; r[13]=(__bf16)f3.y; r[14]=(__bf16)f3.z; r[15]=(__bf16)f3.w;
  return r;
}

// B fragment (32x16 bf16): lane L -> column N = n0 + L%16 (= row of W);
// lanes 0-15 hold K = k0+0..7, k0+16..23; lanes 16-31: +8.
__device__ __forceinline__ v16bf load_fragB(const char* wbuf, int n0, int k0, int lane) {
  const char* p = wbuf + (n0 + (lane & 15)) * ROWB + k0 * 2 + (lane >> 4) * 16;
  v8bf c0 = *(const v8bf*)(p);
  v8bf c1 = *(const v8bf*)(p + 32);
  return __builtin_shufflevector(c0, c1, 0,1,2,3,4,5,6,7,8,9,10,11,12,13,14,15);
}

__device__ __forceinline__ v8f wmma_k128(const v16bf (&a)[4], const char* w,
                                         int n0, int lane) {
  v16bf b0 = load_fragB(w, n0, 0,  lane);
  v16bf b1 = load_fragB(w, n0, 32, lane);
  v16bf b2 = load_fragB(w, n0, 64, lane);
  v16bf b3 = load_fragB(w, n0, 96, lane);
  v8f acc;
#pragma unroll
  for (int e = 0; e < 8; ++e) acc[e] = 0.0f;
  acc = __builtin_amdgcn_wmma_f32_16x16x32_bf16(false, a[0], false, b0, (short)0, acc, false, false);
  acc = __builtin_amdgcn_wmma_f32_16x16x32_bf16(false, a[1], false, b1, (short)0, acc, false, false);
  acc = __builtin_amdgcn_wmma_f32_16x16x32_bf16(false, a[2], false, b2, (short)0, acc, false, false);
  acc = __builtin_amdgcn_wmma_f32_16x16x32_bf16(false, a[3], false, b3, (short)0, acc, false, false);
  return acc;
}

// LIF (subtract reset from PREVIOUS mem, Heaviside spike on new mem),
// spikes stored as bf16 into per-wave LDS buffer at column tile n.
__device__ __forceinline__ void lif_to_lds(v8f& m, v8f cur, char* spkbuf, int n, int lane) {
  const int hi  = lane >> 4;
  const int col = n * 16 + (lane & 15);
  v8f mo = m, mn;
#pragma unroll
  for (int e = 0; e < 8; ++e) {
    float rst = (mo[e] > THR) ? THR : 0.0f;
    mn[e] = BETA * mo[e] + cur[e] - rst;
  }
  m = mn;
#pragma unroll
  for (int e = 0; e < 8; ++e) {
    *(__bf16*)(spkbuf + (e + hi * 8) * ROWB + col * 2) =
        (mn[e] > THR) ? (__bf16)1.0f : (__bf16)0.0f;
  }
}

__global__ void __launch_bounds__(WAVES * 32, 1)
snn_lif3_kernel(const float* __restrict__ x,
                const float* __restrict__ W1,  // [100,128]
                const float* __restrict__ W2,  // [100,100]
                const float* __restrict__ W3,  // [128,100]
                float* __restrict__ out) {
  extern __shared__ char smem[];
  const int tid  = threadIdx.x;
  const int lane = tid & 31;
  const int wid  = tid >> 5;

  // ---- Stage weights as zero-padded bf16 (row-major, 256B row stride) ----
  for (int i = tid; i < HP * 128; i += WAVES * 32) {
    int r = i >> 7, c = i & 127;
    float v = (r < H_REAL) ? W1[r * 128 + c] : 0.0f;
    *(__bf16*)(smem + LDS_W1 + r * ROWB + c * 2) = (__bf16)v;
  }
  for (int i = tid; i < HP * 128; i += WAVES * 32) {
    int r = i >> 7, c = i & 127;
    float v = (r < H_REAL && c < H_REAL) ? W2[r * H_REAL + c] : 0.0f;
    *(__bf16*)(smem + LDS_W2 + r * ROWB + c * 2) = (__bf16)v;
  }
  for (int i = tid; i < 128 * 128; i += WAVES * 32) {
    int r = i >> 7, c = i & 127;
    float v = (c < H_REAL) ? W3[r * H_REAL + c] : 0.0f;
    *(__bf16*)(smem + LDS_W3 + r * ROWB + c * 2) = (__bf16)v;
  }
  // Zero per-wave spike buffers (cols 112..127 must stay 0 forever).
  for (int i = tid; i < (WAVES * WAVE_LDS) / 16; i += WAVES * 32) {
    *(int4*)(smem + LDS_WAVE0 + i * 16) = make_int4(0, 0, 0, 0);
  }
  __syncthreads();

  char* sA = smem + LDS_WAVE0 + wid * WAVE_LDS;  // spk1, bf16 [16][128]
  char* sB = sA + 4096;                          // spk2, bf16 [16][128]

  const int b0 = (blockIdx.x * WAVES + wid) * ROWS_PER_WAVE;

  // Membrane state, WMMA C-layout, lives in VGPRs for all 50 steps.
  v8f m1[NT_H], m2[NT_H], m3[NT_O];
#pragma unroll
  for (int n = 0; n < NT_H; ++n) {
#pragma unroll
    for (int e = 0; e < 8; ++e) { m1[n][e] = 0.0f; m2[n][e] = 0.0f; }
  }
#pragma unroll
  for (int n = 0; n < NT_O; ++n) {
#pragma unroll
    for (int e = 0; e < 8; ++e) m3[n][e] = 0.0f;
  }

  float* spk_out = out;
  float* mem_out = out + (size_t)T_STEPS * BATCH * DIM;

#pragma unroll 1
  for (int t = 0; t < T_STEPS; ++t) {
    const float* xt = x + ((size_t)t * BATCH + b0) * DIM;
    if (t + 1 < T_STEPS) {  // pull next step's x tile toward the caches
      const char* nx = (const char*)(xt + (size_t)BATCH * DIM);
      __builtin_prefetch(nx + lane * 256, 0, 1);
    }

    // ---- Layer 1: A frags straight from global x (f32 -> bf16) ----
    v16bf a1[4];
#pragma unroll
    for (int k = 0; k < 4; ++k) a1[k] = load_fragA_global(xt, k * 32, lane);
#pragma unroll
    for (int n = 0; n < NT_H; ++n) {
      v8f acc = wmma_k128(a1, smem + LDS_W1, n * 16, lane);
      lif_to_lds(m1[n], acc, sA, n, lane);
    }

    // ---- Layer 2: A frags from spk1 LDS, reused across all 7 tiles ----
    v16bf a2[4];
#pragma unroll
    for (int k = 0; k < 4; ++k) a2[k] = load_fragA_lds(sA, k * 32, lane);
#pragma unroll
    for (int n = 0; n < NT_H; ++n) {
      v8f acc = wmma_k128(a2, smem + LDS_W2, n * 16, lane);
      lif_to_lds(m2[n], acc, sB, n, lane);
    }

    // ---- Layer 3: A frags from spk2 LDS; LIF -> spk3, m3 to global ----
    v16bf a3[4];
#pragma unroll
    for (int k = 0; k < 4; ++k) a3[k] = load_fragA_lds(sB, k * 32, lane);
#pragma unroll
    for (int n = 0; n < NT_O; ++n) {
      v8f acc = wmma_k128(a3, smem + LDS_W3, n * 16, lane);
      v8f mo = m3[n], mn;
#pragma unroll
      for (int e = 0; e < 8; ++e) {
        float rst = (mo[e] > THR) ? THR : 0.0f;
        mn[e] = BETA * mo[e] + acc[e] - rst;
      }
      m3[n] = mn;
      // C-layout element (row = e + 8*hi, col = n*16 + lane%16)
      const int hi = lane >> 4;
      size_t base = ((size_t)t * BATCH + b0 + hi * 8) * DIM + n * 16 + (lane & 15);
#pragma unroll
      for (int e = 0; e < 8; ++e) {
        size_t off = base + (size_t)e * DIM;
        spk_out[off] = (mn[e] > THR) ? 1.0f : 0.0f;
        mem_out[off] = mn[e];
      }
    }
  }
}

extern "C" void kernel_launch(void* const* d_in, const int* in_sizes, int n_in,
                              void* d_out, int out_size, void* d_ws, size_t ws_size,
                              hipStream_t stream) {
  (void)in_sizes; (void)n_in; (void)out_size; (void)d_ws; (void)ws_size;
  const float* x  = (const float*)d_in[0];
  const float* W1 = (const float*)d_in[1];
  const float* W2 = (const float*)d_in[2];
  const float* W3 = (const float*)d_in[3];
  float* out = (float*)d_out;

  dim3 grid(BATCH / (ROWS_PER_WAVE * WAVES));  // 256 workgroups
  dim3 block(WAVES * 32);                      // 4 wave32s
  snn_lif3_kernel<<<grid, block, (size_t)LDS_TOTAL, stream>>>(x, W1, W2, W3, out);
}
